// MultiHeadAttention_74801150427830
// MI455X (gfx1250) — matrix operand order repaired and checked by reading, not compiled, not run on hardware
//
#include <hip/hip_runtime.h>

#define N_TOK 4096
#define DIM   512
#define HEADS 8
#define CC    256   // score columns processed per flash iteration
#define BR    32    // rows per workgroup (2 row-tiles of 16)

typedef __bf16 bf16_t;
typedef __attribute__((ext_vector_type(16))) __bf16 v16bf;
typedef __attribute__((ext_vector_type(8)))  __bf16 v8bf;
typedef __attribute__((ext_vector_type(8)))  float  v8f;
typedef unsigned int u32x4 __attribute__((ext_vector_type(4)));
typedef int          i32x8 __attribute__((ext_vector_type(8)));
typedef int          i32x4 __attribute__((ext_vector_type(4)));

__device__ __forceinline__ unsigned short f32_to_bf16(float f) {
  union { float f; unsigned u; } c; c.f = f;
  unsigned u = c.u;
  u += 0x7FFFu + ((u >> 16) & 1u);          // round-to-nearest-even
  return (unsigned short)(u >> 16);
}

// A-matrix (16x32 bf16, MxK): lane holds row lane%16; K chunks at
// 8*(lane/16) and 16+8*(lane/16). src already offset to (row_base, k_base).
__device__ __forceinline__ v16bf load_a16(const bf16_t* src, int ld, int lane) {
  const bf16_t* p = src + (size_t)(lane & 15) * ld + ((lane >> 4) << 3);
  v8bf lo = *reinterpret_cast<const v8bf*>(p);
  v8bf hi = *reinterpret_cast<const v8bf*>(p + 16);
  v16bf r;
#pragma unroll
  for (int i = 0; i < 8; ++i) { r[i] = lo[i]; r[i + 8] = hi[i]; }
  return r;
}

// B-matrix (32x16 bf16, KxN) where B[k][n] = src[n][k] (src row-major):
// lane holds column lane%16; K chunks at 8*(lane/16) and 16+8*(lane/16).
__device__ __forceinline__ v16bf load_bT16(const bf16_t* src, int ld, int lane) {
  const bf16_t* p = src + (size_t)(lane & 15) * ld + ((lane >> 4) << 3);
  v8bf lo = *reinterpret_cast<const v8bf*>(p);
  v8bf hi = *reinterpret_cast<const v8bf*>(p + 16);
  v16bf r;
#pragma unroll
  for (int i = 0; i < 8; ++i) { r[i] = lo[i]; r[i + 8] = hi[i]; }
  return r;
}

__device__ __forceinline__ v8f wmma_bf16(v16bf a, v16bf b, v8f c) {
  return __builtin_amdgcn_wmma_f32_16x16x32_bf16(
      false, a, false, b, (short)0, c, false, false);
}

// ---------------------------------------------------------------------------
// Kernel 0: f32 -> bf16 conversion (x, x^T, W)
// ---------------------------------------------------------------------------
__global__ void convert_kernel(const float* __restrict__ x,
                               const float* __restrict__ W,
                               unsigned short* __restrict__ x_b,
                               unsigned short* __restrict__ xT_b,
                               unsigned short* __restrict__ W_b) {
  int tid = blockIdx.x * blockDim.x + threadIdx.x;
  int stride = gridDim.x * blockDim.x;
  for (int i = tid; i < N_TOK * DIM; i += stride) {
    unsigned short v = f32_to_bf16(x[i]);
    x_b[i] = v;
    int n = i / DIM, d = i % DIM;
    xT_b[(size_t)d * N_TOK + n] = v;
  }
  for (int i = tid; i < HEADS * DIM * DIM; i += stride)
    W_b[i] = f32_to_bf16(W[i]);
}

// ---------------------------------------------------------------------------
// Kernel 1: h[hd] = x @ W[hd]^T + b[hd], stored bf16. One 16x16 tile / wave.
// ---------------------------------------------------------------------------
__global__ void proj_kernel(const unsigned short* __restrict__ x_b,
                            const unsigned short* __restrict__ W_b,
                            const float* __restrict__ bias,
                            unsigned short* __restrict__ h_b) {
  const int w = threadIdx.x >> 5, lane = threadIdx.x & 31;
  const int ct = DIM / 16;     // 32 col tiles
  const int rt = N_TOK / 16;   // 256 row tiles
  int tile = blockIdx.x * (blockDim.x >> 5) + w;
  int hd = tile / (rt * ct);
  int rc = tile % (rt * ct);
  int rb = (rc / ct) * 16;     // row base in N
  int ob = (rc % ct) * 16;     // output-feature base in D

  const bf16_t* xb = reinterpret_cast<const bf16_t*>(x_b);
  const bf16_t* Wb = reinterpret_cast<const bf16_t*>(W_b) + (size_t)hd * DIM * DIM;

  v8f acc = {};
#pragma unroll 4
  for (int k = 0; k < DIM; k += 32) {
    v16bf a = load_a16(xb + (size_t)rb * DIM + k, DIM, lane);   // A[m][k]=x[rb+m][k]
    v16bf b = load_bT16(Wb + (size_t)ob * DIM + k, DIM, lane);  // B[k][n]=W[ob+n][k]
    acc = wmma_bf16(a, b, acc);
  }
  const int m0 = (lane >> 4) << 3;
  const int n  = lane & 15;
  float bv = bias[hd * DIM + ob + n];
  unsigned short* hp = h_b + ((size_t)hd * N_TOK + rb) * DIM + ob + n;
#pragma unroll
  for (int j = 0; j < 8; ++j)
    hp[(size_t)(m0 + j) * DIM] = f32_to_bf16(acc[j] + bv);
}

// ---------------------------------------------------------------------------
// Kernel 2: flash attention. WG = (head, 32-row block), 8 waves.
//   phase A : wave w -> score tiles S[32, 32] for its columns (K = 512)
//   softmax : online max/sum via register-layout shuffles + LDS stats
//   phase B : wave w -> acc[32, 64] += P[32,256] @ x[256, 64-slice]
// ---------------------------------------------------------------------------
__global__ void attn_kernel(const unsigned short* __restrict__ h_b,
                            const unsigned short* __restrict__ xT_b,
                            float* __restrict__ out) {
  __shared__ __align__(16) unsigned short lds_h[BR * DIM];  // 32 KB: h row block
  __shared__ __align__(16) unsigned short lds_p[BR * CC];   // 16 KB: exp'd probs
  __shared__ float wmax[8 * BR];    // per-wave row maxima
  __shared__ float wsum[8 * BR];    // per-wave row partial sums
  __shared__ float row_m[BR], row_l[BR], row_scale[BR];

  const int w = threadIdx.x >> 5, lane = threadIdx.x & 31;
  const int hd = blockIdx.x / (N_TOK / BR);
  const int rb = (blockIdx.x % (N_TOK / BR)) * BR;
  const unsigned short* hhead = h_b + (size_t)hd * N_TOK * DIM;
  const bf16_t* hheadb = reinterpret_cast<const bf16_t*>(hhead);
  const bf16_t* ldsh = reinterpret_cast<const bf16_t*>(lds_h);
  const bf16_t* ldsp = reinterpret_cast<const bf16_t*>(lds_p);
  const bf16_t* xT = reinterpret_cast<const bf16_t*>(xT_b);

  // ---- TDM: stage the 32x512 bf16 h row block (32 KB contiguous) into LDS
  if (threadIdx.x == 0) {
    unsigned long long gaddr =
        (unsigned long long)(uintptr_t)(hhead + (size_t)rb * DIM);
    unsigned lds_off = (unsigned)(uintptr_t)(&lds_h[0]);
    u32x4 g0;
    g0[0] = 1u;                                   // count=1, user mode
    g0[1] = lds_off;                              // lds_addr
    g0[2] = (unsigned)(gaddr & 0xFFFFFFFFull);    // global_addr[31:0]
    g0[3] = (unsigned)(gaddr >> 32) | (2u << 30); // global_addr[56:32], type=2
    i32x8 g1;
    g1[0] = (int)(3u << 16);        // data_size = 8 bytes
    g1[1] = (int)(0x1000u << 16);   // tensor_dim0 = 4096 (8B units), low 16
    g1[2] = (int)(1u << 16);        // tensor_dim0 hi = 0; tensor_dim1 = 1
    g1[3] = (int)(0x1000u << 16);   // tile_dim0 = 4096
    g1[4] = 0;                      // tile_dim1 = tile_dim2 = 0 (1-D tile)
    g1[5] = 4096;                   // tensor_dim0_stride
    g1[6] = 0;
    g1[7] = 0;
    i32x4 g2; g2[0] = 1; g2[1] = 0; g2[2] = 0; g2[3] = 0;
    i32x4 g3; g3[0] = 0; g3[1] = 0; g3[2] = 0; g3[3] = 0;
    i32x8 gz; gz[0]=0; gz[1]=0; gz[2]=0; gz[3]=0; gz[4]=0; gz[5]=0; gz[6]=0; gz[7]=0;
    __builtin_amdgcn_tensor_load_to_lds(g0, g1, g2, g3, gz, 0);
    __builtin_amdgcn_s_wait_tensorcnt(0);
  }
  if (threadIdx.x < BR) { row_m[threadIdx.x] = -INFINITY; row_l[threadIdx.x] = 0.f; }
  __syncthreads();

  v8f acc[8] = {};                 // 32 x 64 f32 output slice per wave
  const int dcol = w * 64;
  const int m0 = (lane >> 4) << 3;
  const int nn = lane & 15;

  for (int cc = 0; cc < N_TOK; cc += CC) {
    // ---- phase A: scores S[2][2] tiles, K = 512; B reused across 2 row tiles
    v8f s[2][2] = {};
    const int cbase = cc + w * 32;
#pragma unroll 2
    for (int k = 0; k < DIM; k += 32) {
      v16bf a0 = load_a16(ldsh + k, DIM, lane);
      v16bf a1 = load_a16(ldsh + (size_t)16 * DIM + k, DIM, lane);
      v16bf b0 = load_bT16(hheadb + (size_t)cbase * DIM + k, DIM, lane);
      v16bf b1 = load_bT16(hheadb + (size_t)(cbase + 16) * DIM + k, DIM, lane);
      s[0][0] = wmma_bf16(a0, b0, s[0][0]);
      s[0][1] = wmma_bf16(a0, b1, s[0][1]);
      s[1][0] = wmma_bf16(a1, b0, s[1][0]);
      s[1][1] = wmma_bf16(a1, b1, s[1][1]);
    }

    // per-wave row maxima over this wave's 32 columns (C layout: row=rt*16+m0+j)
#pragma unroll
    for (int rt = 0; rt < 2; ++rt)
#pragma unroll
      for (int j = 0; j < 8; ++j) {
        float mx = fmaxf(s[rt][0][j], s[rt][1][j]);
#pragma unroll
        for (int off = 8; off >= 1; off >>= 1)
          mx = fmaxf(mx, __shfl_xor(mx, off, 32));   // stays in 16-lane half
        if (nn == 0) wmax[w * BR + rt * 16 + m0 + j] = mx;
      }
    __syncthreads();   // B1

    // ---- combine stats: wave w owns rows w*4 .. w*4+3
#pragma unroll
    for (int rr = 0; rr < 4; ++rr) {
      const int r = w * 4 + rr;
      float mc = -INFINITY;
#pragma unroll
      for (int ww = 0; ww < 8; ++ww) mc = fmaxf(mc, wmax[ww * BR + r]);
      float mold = row_m[r];
      float mnew = fmaxf(mold, mc);
      if (lane == 0) {
        row_m[r] = mnew;
        row_scale[r] = __expf(mold - mnew);
      }
    }
    __syncthreads();   // B2

    // ---- P = exp(S - m_new); write bf16 P to LDS; per-wave row sums; rescale acc
#pragma unroll
    for (int rt = 0; rt < 2; ++rt)
#pragma unroll
      for (int j = 0; j < 8; ++j) {
        const int r = rt * 16 + m0 + j;
        float mr = row_m[r];
        float sc = row_scale[r];
        float p0 = __expf(s[rt][0][j] - mr);
        float p1 = __expf(s[rt][1][j] - mr);
        lds_p[r * CC + w * 32 + nn]      = f32_to_bf16(p0);
        lds_p[r * CC + w * 32 + 16 + nn] = f32_to_bf16(p1);
        float ps = p0 + p1;
#pragma unroll
        for (int off = 8; off >= 1; off >>= 1)
          ps += __shfl_xor(ps, off, 32);
        if (nn == 0) wsum[w * BR + r] = ps;
        acc[rt * 4 + 0][j] *= sc;
        acc[rt * 4 + 1][j] *= sc;
        acc[rt * 4 + 2][j] *= sc;
        acc[rt * 4 + 3][j] *= sc;
      }
    __syncthreads();   // B3

    // ---- owner waves fold sums into running denominators
#pragma unroll
    for (int rr = 0; rr < 4; ++rr) {
      const int r = w * 4 + rr;
      if (lane == 0) {
        float sacc = 0.f;
#pragma unroll
        for (int ww = 0; ww < 8; ++ww) sacc += wsum[ww * BR + r];
        row_l[r] = row_l[r] * row_scale[r] + sacc;
      }
    }

    // ---- phase B: acc += P[32,256] @ x[cc..cc+256][dcol..dcol+64]
#pragma unroll 2
    for (int kk = 0; kk < CC; kk += 32) {
      v16bf a0 = load_a16(ldsp + kk, CC, lane);
      v16bf a1 = load_a16(ldsp + (size_t)16 * CC + kk, CC, lane);
#pragma unroll
      for (int t = 0; t < 4; ++t) {
        v16bf b = load_bT16(xT + (size_t)(dcol + t * 16) * N_TOK + cc + kk,
                            N_TOK, lane);
        acc[t]     = wmma_bf16(a0, b, acc[t]);
        acc[4 + t] = wmma_bf16(a1, b, acc[4 + t]);
      }
    }
    // next iteration's B1 guards wmax / lds_p reuse
  }
  __syncthreads();   // ensure all row_l updates are visible

  // ---- epilogue: divide by row sums, write out[n][hd*DIM + d]
#pragma unroll
  for (int rt = 0; rt < 2; ++rt)
#pragma unroll
    for (int j = 0; j < 8; ++j) {
      float inv = 1.0f / row_l[rt * 16 + m0 + j];
      float* op = out + (size_t)(rb + rt * 16 + m0 + j) * (HEADS * DIM) + hd * DIM;
#pragma unroll
      for (int t = 0; t < 4; ++t)
        op[dcol + t * 16 + nn] = acc[rt * 4 + t][j] * inv;
    }
}

// ---------------------------------------------------------------------------
extern "C" void kernel_launch(void* const* d_in, const int* in_sizes, int n_in,
                              void* d_out, int out_size, void* d_ws, size_t ws_size,
                              hipStream_t stream) {
  const float* x = (const float*)d_in[0];   // [N, D]
  const float* W = (const float*)d_in[1];   // [H, D, D]
  const float* b = (const float*)d_in[2];   // [H, D]
  float* out = (float*)d_out;               // [N, H*D]

  // workspace carve-up (bf16 buffers): 4 + 4 + 4 + 32 = 44 MB
  unsigned short* x_b  = (unsigned short*)d_ws;
  unsigned short* xT_b = x_b  + (size_t)N_TOK * DIM;
  unsigned short* W_b  = xT_b + (size_t)DIM * N_TOK;
  unsigned short* h_b  = W_b  + (size_t)HEADS * DIM * DIM;

  convert_kernel<<<512, 256, 0, stream>>>(x, W, x_b, xT_b, W_b);

  int proj_tiles = HEADS * (N_TOK / 16) * (DIM / 16);   // 65536 tiles
  proj_kernel<<<proj_tiles / 8, 256, 0, stream>>>(x_b, W_b, b, h_b);

  attn_kernel<<<HEADS * (N_TOK / BR), 256, 0, stream>>>(h_b, xT_b, out);
}
